// SeqTransformerEncoderLayer_24215025614962
// MI455X (gfx1250) — matrix operand order repaired and checked by reading, not compiled, not run on hardware
//
#include <hip/hip_runtime.h>

#define B_ 4
#define S_ 2048
#define D_ 512
#define H_ 8
#define HD_ 64
#define DFF_ 1024
#define EPS_ 1e-5f

typedef __bf16 bf16_t;
typedef __attribute__((ext_vector_type(16))) bf16_t v16bf;
typedef __attribute__((ext_vector_type(8))) float v8f;
typedef unsigned short u16;

union Frag {
  v16bf v;
  unsigned u[8];
  uint4 q[2];
};

__device__ __forceinline__ u16 f2bf(float f) {
  bf16_t h = (bf16_t)f;
  union { bf16_t b; u16 u; } cv;
  cv.b = h;
  return cv.u;
}

// -------------------------------------------------------------------------
// fp32 -> packed bf16 conversion (one pass, amortized outside GEMM loops)
// -------------------------------------------------------------------------
__global__ __launch_bounds__(256) void cvt_f32_bf16(
    const float* __restrict__ in, u16* __restrict__ out, int n2)
{
  int i = blockIdx.x * 256 + threadIdx.x;
  if (i < n2) {
    float2 f = ((const float2*)in)[i];
    unsigned d = (unsigned)f2bf(f.x) | ((unsigned)f2bf(f.y) << 16);
    ((unsigned*)out)[i] = d;
  }
}

// -------------------------------------------------------------------------
// Y = X[M,K](bf16) @ W[N,K](bf16)^T + bias, optional ReLU.
// 256 threads = 8 waves (4xM, 2xN); block tile 128x128; wave tile 32x64:
// 2x4 WMMA accumulators -> 8 WMMA per 12 b128 fragment loads per k-step.
// Outputs (each optional): Yf f32 [M,N], Yh bf16 [M,N], Yt bf16 [N,M].
// -------------------------------------------------------------------------
__global__ __launch_bounds__(256) void gemm_wmma_bf16(
    const u16* __restrict__ X, const u16* __restrict__ W,
    const float* __restrict__ bias,
    float* __restrict__ Yf, u16* __restrict__ Yh, u16* __restrict__ Yt,
    int M, int N, int K, int relu)
{
  const int lane = threadIdx.x & 31;
  const int wave = threadIdx.x >> 5;
  const int wm = wave & 3;   // 4 waves in M
  const int wn = wave >> 2;  // 2 waves in N
  const int col = lane & 15;
  const int half = lane >> 4;

  const int rowBase = blockIdx.y * 128 + wm * 32;
  const int colBase = blockIdx.x * 128 + wn * 64;

  v8f zero = {};
  v8f acc[2][4];
#pragma unroll
  for (int i = 0; i < 2; ++i)
#pragma unroll
    for (int j = 0; j < 4; ++j) acc[i][j] = zero;

  const u16* Xr0 = X + (size_t)(rowBase + col) * K;
  const u16* Xr1 = X + (size_t)(rowBase + 16 + col) * K;
  const u16* Wr[4];
#pragma unroll
  for (int j = 0; j < 4; ++j) Wr[j] = W + (size_t)(colBase + j * 16 + col) * K;

  for (int k0 = 0; k0 < K; k0 += 32) {
    Frag a0, a1, b[4];
    const int ka = k0 + half * 8;   // A: u[0..3]=pairs ka..ka+7, u[4..7]=ka+16..ka+23
    const int kb = k0 + half * 8;   // B: u[0..3]=pairs kb..kb+7, u[4..7]=kb+16..kb+23
    a0.q[0] = *(const uint4*)(Xr0 + ka);
    a0.q[1] = *(const uint4*)(Xr0 + ka + 16);
    a1.q[0] = *(const uint4*)(Xr1 + ka);
    a1.q[1] = *(const uint4*)(Xr1 + ka + 16);
#pragma unroll
    for (int j = 0; j < 4; ++j) {
      b[j].q[0] = *(const uint4*)(Wr[j] + kb);
      b[j].q[1] = *(const uint4*)(Wr[j] + kb + 16);
    }
#pragma unroll
    for (int j = 0; j < 4; ++j) {
      acc[0][j] = __builtin_amdgcn_wmma_f32_16x16x32_bf16(false, a0.v, false, b[j].v, (short)0, acc[0][j], false, false);
      acc[1][j] = __builtin_amdgcn_wmma_f32_16x16x32_bf16(false, a1.v, false, b[j].v, (short)0, acc[1][j], false, false);
    }
  }

#pragma unroll
  for (int i = 0; i < 2; ++i) {
    const int r0 = rowBase + i * 16 + half * 8;
#pragma unroll
    for (int j = 0; j < 4; ++j) {
      const int c = colBase + j * 16 + col;
      const float bb = bias[c];
      float vals[8];
#pragma unroll
      for (int r = 0; r < 8; ++r) {
        float v = acc[i][j][r] + bb;
        if (relu) v = fmaxf(v, 0.f);
        vals[r] = v;
      }
      if (Yf) {
#pragma unroll
        for (int r = 0; r < 8; ++r) Yf[(size_t)(r0 + r) * N + c] = vals[r];
      }
      if (Yh) {
#pragma unroll
        for (int r = 0; r < 8; ++r) Yh[(size_t)(r0 + r) * N + c] = f2bf(vals[r]);
      }
      if (Yt) {  // transposed bf16: 8 consecutive rows -> one b128 store
        uint4 pk;
        pk.x = (unsigned)f2bf(vals[0]) | ((unsigned)f2bf(vals[1]) << 16);
        pk.y = (unsigned)f2bf(vals[2]) | ((unsigned)f2bf(vals[3]) << 16);
        pk.z = (unsigned)f2bf(vals[4]) | ((unsigned)f2bf(vals[5]) << 16);
        pk.w = (unsigned)f2bf(vals[6]) | ((unsigned)f2bf(vals[7]) << 16);
        *(uint4*)(Yt + (size_t)c * M + r0) = pk;
      }
    }
  }
}

// -------------------------------------------------------------------------
// Flash attention: O = softmax(Q K^T / 8) V, one (b,h) per blockIdx.y.
// Q,K bf16 [B,S,D] (head h in cols h*64..); V bf16 transposed [D][M].
// Block: 128 threads = 4 waves; wave owns 16 query rows; key blocks of 64.
// -------------------------------------------------------------------------
__global__ __launch_bounds__(128) void attn_wmma(
    const u16* __restrict__ qh, const u16* __restrict__ kh,
    const u16* __restrict__ vt, u16* __restrict__ oh)
{
  __shared__ u16 plds[4][16 * 64];  // per-wave P tile (bf16, row-major)

  const int lane = threadIdx.x & 31;
  const int wave = threadIdx.x >> 5;
  const int col = lane & 15;
  const int half = lane >> 4;
  const int b = blockIdx.y / H_;
  const int h = blockIdx.y % H_;
  const int qr0 = blockIdx.x * 64 + wave * 16;

  const size_t baseOff = (size_t)b * S_ * D_ + (size_t)h * HD_;
  const u16* Q = qh + baseOff;
  const u16* Kp = kh + baseOff;
  const int M = B_ * S_;
  const u16* Vt = vt + (size_t)h * HD_ * M + (size_t)b * S_;
  u16* O = oh + baseOff;

  // Q A-fragments are loop-invariant
  Frag aq[2];
  {
    const u16* Qr = Q + (size_t)(qr0 + col) * D_;
#pragma unroll
    for (int ks = 0; ks < 2; ++ks) {
      aq[ks].q[0] = *(const uint4*)(Qr + ks * 32 + half * 8);
      aq[ks].q[1] = *(const uint4*)(Qr + ks * 32 + half * 8 + 16);
    }
  }

  v8f zero = {};
  v8f accO[4];
#pragma unroll
  for (int nt = 0; nt < 4; ++nt) accO[nt] = zero;
  float mrow[8], lrow[8];
#pragma unroll
  for (int r = 0; r < 8; ++r) { mrow[r] = -1e30f; lrow[r] = 0.f; }

  for (int kb = 0; kb < S_; kb += 64) {
    // ---- S = Q (16x64) x K_block^T (64x64): 4 N-tiles x 2 k-steps
    v8f sc[4];
#pragma unroll
    for (int nt = 0; nt < 4; ++nt) {
      const u16* Kr = Kp + (size_t)(kb + nt * 16 + col) * D_;
      Frag bk0, bk1;
      bk0.q[0] = *(const uint4*)(Kr + half * 8);
      bk0.q[1] = *(const uint4*)(Kr + half * 8 + 16);
      bk1.q[0] = *(const uint4*)(Kr + half * 8 + 32);
      bk1.q[1] = *(const uint4*)(Kr + half * 8 + 48);
      v8f s = __builtin_amdgcn_wmma_f32_16x16x32_bf16(false, aq[0].v, false, bk0.v, (short)0, zero, false, false);
      s = __builtin_amdgcn_wmma_f32_16x16x32_bf16(false, aq[1].v, false, bk1.v, (short)0, s, false, false);
      sc[nt] = s;
    }
    // ---- online softmax; row lives in a 16-lane half group
#pragma unroll
    for (int r = 0; r < 8; ++r) {
      float mx = -1e30f;
#pragma unroll
      for (int nt = 0; nt < 4; ++nt) { sc[nt][r] *= 0.125f; mx = fmaxf(mx, sc[nt][r]); }
      mx = fmaxf(mx, __shfl_xor(mx, 1, 32));
      mx = fmaxf(mx, __shfl_xor(mx, 2, 32));
      mx = fmaxf(mx, __shfl_xor(mx, 4, 32));
      mx = fmaxf(mx, __shfl_xor(mx, 8, 32));
      float mnew = fmaxf(mrow[r], mx);
      float alpha = __expf(mrow[r] - mnew);
      mrow[r] = mnew;
      float sum = 0.f;
#pragma unroll
      for (int nt = 0; nt < 4; ++nt) {
        float p = __expf(sc[nt][r] - mnew);
        sc[nt][r] = p;
        sum += p;
      }
      sum += __shfl_xor(sum, 1, 32);
      sum += __shfl_xor(sum, 2, 32);
      sum += __shfl_xor(sum, 4, 32);
      sum += __shfl_xor(sum, 8, 32);
      lrow[r] = lrow[r] * alpha + sum;
#pragma unroll
      for (int nt = 0; nt < 4; ++nt) accO[nt][r] *= alpha;
    }
    // ---- stage P: C layout -> LDS row-major bf16 (16x64)
#pragma unroll
    for (int nt = 0; nt < 4; ++nt)
#pragma unroll
      for (int r = 0; r < 8; ++r)
        plds[wave][(r + 8 * half) * 64 + nt * 16 + col] = f2bf(sc[nt][r]);
    __syncthreads();
    // ---- reload P as A-fragments (ds_load_b128)
    Frag ap[2];
#pragma unroll
    for (int ks = 0; ks < 2; ++ks) {
      ap[ks].q[0] = *(const uint4*)&plds[wave][col * 64 + ks * 32 + half * 8];
      ap[ks].q[1] = *(const uint4*)&plds[wave][col * 64 + ks * 32 + half * 8 + 16];
    }
    // ---- O += P (16x64) x V_block (64x64); V^T rows are contiguous in k
#pragma unroll
    for (int nt = 0; nt < 4; ++nt) {
#pragma unroll
      for (int ks = 0; ks < 2; ++ks) {
        const u16* Vr = Vt + (size_t)(nt * 16 + col) * M + kb + ks * 32 + half * 8;
        Frag bv;
        bv.q[0] = *(const uint4*)(Vr);
        bv.q[1] = *(const uint4*)(Vr + 16);
        accO[nt] = __builtin_amdgcn_wmma_f32_16x16x32_bf16(false, ap[ks].v, false, bv.v, (short)0, accO[nt], false, false);
      }
    }
    __syncthreads();
  }
  // ---- normalize and store (bf16, feeds the out-projection GEMM)
#pragma unroll
  for (int nt = 0; nt < 4; ++nt)
#pragma unroll
    for (int r = 0; r < 8; ++r)
      O[(size_t)(qr0 + r + 8 * half) * D_ + nt * 16 + col] = f2bf(accO[nt][r] / lrow[r]);
}

// -------------------------------------------------------------------------
// out = LayerNorm(x + y)*g + beta ; optional bf16 copy of the output.
// One 256-thread block per 512-wide row.
// -------------------------------------------------------------------------
__global__ __launch_bounds__(256) void add_ln_kernel(
    const float* __restrict__ x, const float* __restrict__ y,
    const float* __restrict__ g, const float* __restrict__ beta,
    float* __restrict__ out, u16* __restrict__ outh)
{
  const int row = blockIdx.x;
  const int tid = threadIdx.x;
  const float* xr = x + (size_t)row * D_;
  const float* yr = y + (size_t)row * D_;
  float v0 = xr[tid] + yr[tid];
  float v1 = xr[tid + 256] + yr[tid + 256];
  float s = v0 + v1;
  float s2 = v0 * v0 + v1 * v1;
#pragma unroll
  for (int off = 16; off > 0; off >>= 1) {
    s += __shfl_xor(s, off, 32);
    s2 += __shfl_xor(s2, off, 32);
  }
  __shared__ float sh0[8], sh1[8];
  int lane = tid & 31, wv = tid >> 5;
  if (lane == 0) { sh0[wv] = s; sh1[wv] = s2; }
  __syncthreads();
  if (wv == 0) {
    float a = (lane < 8) ? sh0[lane] : 0.f;
    float c = (lane < 8) ? sh1[lane] : 0.f;
#pragma unroll
    for (int off = 4; off > 0; off >>= 1) {
      a += __shfl_xor(a, off, 32);
      c += __shfl_xor(c, off, 32);
    }
    if (lane == 0) { sh0[0] = a; sh1[0] = c; }
  }
  __syncthreads();
  float mean = sh0[0] * (1.f / D_);
  float var = sh1[0] * (1.f / D_) - mean * mean;
  float rstd = rsqrtf(var + EPS_);
  float o0 = (v0 - mean) * rstd * g[tid] + beta[tid];
  float o1 = (v1 - mean) * rstd * g[tid + 256] + beta[tid + 256];
  out[(size_t)row * D_ + tid] = o0;
  out[(size_t)row * D_ + tid + 256] = o1;
  if (outh) {
    outh[(size_t)row * D_ + tid] = f2bf(o0);
    outh[(size_t)row * D_ + tid + 256] = f2bf(o1);
  }
}

extern "C" void kernel_launch(void* const* d_in, const int* in_sizes, int n_in,
                              void* d_out, int out_size, void* d_ws, size_t ws_size,
                              hipStream_t stream) {
  (void)in_sizes; (void)n_in; (void)out_size; (void)ws_size;
  const float* x   = (const float*)d_in[0];
  const float* wq  = (const float*)d_in[1];
  const float* bq  = (const float*)d_in[2];
  const float* wk  = (const float*)d_in[3];
  const float* bk  = (const float*)d_in[4];
  const float* wv  = (const float*)d_in[5];
  const float* bv  = (const float*)d_in[6];
  const float* wo  = (const float*)d_in[7];
  const float* bo  = (const float*)d_in[8];
  const float* w1  = (const float*)d_in[9];
  const float* b1  = (const float*)d_in[10];
  const float* w2  = (const float*)d_in[11];
  const float* b2  = (const float*)d_in[12];
  const float* g1  = (const float*)d_in[13];
  const float* be1 = (const float*)d_in[14];
  const float* g2  = (const float*)d_in[15];
  const float* be2 = (const float*)d_in[16];
  float* out = (float*)d_out;

  const int M = B_ * S_;
  const size_t BSD = (size_t)M * D_;
  const size_t DD = (size_t)D_ * D_;
  const size_t DDF = (size_t)D_ * DFF_;

  // workspace carve-up (256B aligned chunks)
  char* base = (char*)d_ws;
  size_t off = 0;
  auto alloc = [&](size_t bytes) -> void* {
    void* p = base + off;
    off = (off + bytes + 255) & ~(size_t)255;
    return p;
  };
  u16* xh   = (u16*)alloc(BSD * 2);
  u16* wqh  = (u16*)alloc(DD * 2);
  u16* wkh  = (u16*)alloc(DD * 2);
  u16* wvh  = (u16*)alloc(DD * 2);
  u16* woh  = (u16*)alloc(DD * 2);
  u16* w1h  = (u16*)alloc(DDF * 2);
  u16* w2h  = (u16*)alloc(DDF * 2);
  u16* qhb  = (u16*)alloc(BSD * 2);
  u16* khb  = (u16*)alloc(BSD * 2);
  u16* vtb  = (u16*)alloc(BSD * 2);      // V^T: [D][M]
  u16* avh  = (u16*)alloc(BSD * 2);
  u16* x1h  = (u16*)alloc(BSD * 2);
  u16* hh   = (u16*)alloc((size_t)M * DFF_ * 2);
  float* t0 = (float*)alloc(BSD * 4);    // attn_out f32, later reused for ff
  float* x1f = (float*)alloc(BSD * 4);

  dim3 blk(256);
  // fp32 -> bf16 one-shot conversions
  cvt_f32_bf16<<<dim3((unsigned)(BSD / 2 / 256)), blk, 0, stream>>>(x, xh, (int)(BSD / 2));
  cvt_f32_bf16<<<dim3((unsigned)(DD / 2 / 256)), blk, 0, stream>>>(wq, wqh, (int)(DD / 2));
  cvt_f32_bf16<<<dim3((unsigned)(DD / 2 / 256)), blk, 0, stream>>>(wk, wkh, (int)(DD / 2));
  cvt_f32_bf16<<<dim3((unsigned)(DD / 2 / 256)), blk, 0, stream>>>(wv, wvh, (int)(DD / 2));
  cvt_f32_bf16<<<dim3((unsigned)(DD / 2 / 256)), blk, 0, stream>>>(wo, woh, (int)(DD / 2));
  cvt_f32_bf16<<<dim3((unsigned)(DDF / 2 / 256)), blk, 0, stream>>>(w1, w1h, (int)(DDF / 2));
  cvt_f32_bf16<<<dim3((unsigned)(DDF / 2 / 256)), blk, 0, stream>>>(w2, w2h, (int)(DDF / 2));

  dim3 gN512(512 / 128, M / 128);
  dim3 gN1024(DFF_ / 128, M / 128);

  // QKV projections (Q,K bf16 row-major; V bf16 transposed)
  gemm_wmma_bf16<<<gN512, blk, 0, stream>>>(xh, wqh, bq, nullptr, qhb, nullptr, M, 512, 512, 0);
  gemm_wmma_bf16<<<gN512, blk, 0, stream>>>(xh, wkh, bk, nullptr, khb, nullptr, M, 512, 512, 0);
  gemm_wmma_bf16<<<gN512, blk, 0, stream>>>(xh, wvh, bv, nullptr, nullptr, vtb, M, 512, 512, 0);

  // flash attention
  attn_wmma<<<dim3(S_ / 64, B_ * H_), dim3(128), 0, stream>>>(qhb, khb, vtb, avh);

  // output projection + residual LN (x1 kept in f32 and bf16)
  gemm_wmma_bf16<<<gN512, blk, 0, stream>>>(avh, woh, bo, t0, nullptr, nullptr, M, 512, 512, 0);
  add_ln_kernel<<<dim3(M), blk, 0, stream>>>(x, t0, g1, be1, x1f, x1h);

  // FFN
  gemm_wmma_bf16<<<gN1024, blk, 0, stream>>>(x1h, w1h, b1, nullptr, hh, nullptr, M, DFF_, 512, 1);
  gemm_wmma_bf16<<<gN512, blk, 0, stream>>>(hh, w2h, b2, t0, nullptr, nullptr, M, 512, DFF_, 0);
  add_ln_kernel<<<dim3(M), blk, 0, stream>>>(x1f, t0, g2, be2, out, nullptr);
}
